// GNNQNetwork_35785667510417
// MI455X (gfx1250) — matrix operand order repaired and checked by reading, not compiled, not run on hardware
//
#include <hip/hip_runtime.h>
#include <hip/hip_bf16.h>

#define N_NODES 65536
#define HIDF    128
#define NHEAD   4
#define CH      32
#define BG      64
#define NPG     1024
#define AACT    32
#define E_DIR   262144
#define E_TOT   (E_DIR + N_NODES)   // 327680 (self loops appended)
#define LLAYERS 3
#define MLP_IN  515                  // 4*128 + 3
#define NEG_SLOPE 0.2f

typedef _Float16 half8  __attribute__((ext_vector_type(8)));
typedef _Float16 half16 __attribute__((ext_vector_type(16)));
typedef float    f32x8  __attribute__((ext_vector_type(8)));

// ---------------- weight prep: Wl/Wr -> transposed f16 [l][n][k] ----------------
__global__ void k_prep_w(const float* __restrict__ gWl, const float* __restrict__ gWr,
                         _Float16* __restrict__ WlT, _Float16* __restrict__ WrT) {
    int idx = blockIdx.x * 256 + threadIdx.x;          // 3*128*128 = 49152
    if (idx >= LLAYERS * HIDF * HIDF) return;
    int l = idx >> 14, nk = idx & 16383;
    int n = nk >> 7, k = nk & 127;
    WlT[idx] = (_Float16)gWl[(l << 14) + (k << 7) + n];
    WrT[idx] = (_Float16)gWr[(l << 14) + (k << 7) + n];
}

// ---------------- node encoder: h = relu(x*Wn + bn), NF==1 ----------------
__global__ void k_node_encode(const float* __restrict__ x, const float* __restrict__ Wn,
                              const float* __restrict__ bn,
                              float* __restrict__ h, _Float16* __restrict__ hh) {
    size_t idx = (size_t)blockIdx.x * 256 + threadIdx.x;   // N*HID
    if (idx >= (size_t)N_NODES * HIDF) return;
    int f = (int)(idx & 127);
    size_t i = idx >> 7;
    float v = fmaxf(fmaf(x[i], Wn[f], bn[f]), 0.f);
    h[idx] = v;
    hh[idx] = (_Float16)v;
}

// ---------------- edge-attr mean (for self-loop fill) ----------------
__global__ void k_zero_scalar(float* p) { if (blockIdx.x == 0 && threadIdx.x == 0) p[0] = 0.f; }

__global__ void k_edge_sum(const float* __restrict__ ea, float* __restrict__ out) {
    __shared__ float red[256];
    size_t idx = (size_t)blockIdx.x * 256 + threadIdx.x;
    red[threadIdx.x] = (idx < E_DIR) ? ea[idx] : 0.f;
    __syncthreads();
    for (int s = 128; s > 0; s >>= 1) {
        if (threadIdx.x < s) red[threadIdx.x] += red[threadIdx.x + s];
        __syncthreads();
    }
    if (threadIdx.x == 0) atomicAdd(out, red[0]);
}

// ---------------- CSR-by-destination build (once per launch) ----------------
__global__ void k_zero_deg(int* __restrict__ deg) {
    int i = blockIdx.x * 256 + threadIdx.x;
    if (i < N_NODES) deg[i] = 0;
}

__global__ void k_hist(const int* __restrict__ dst, int* __restrict__ deg) {
    int e = blockIdx.x * 256 + threadIdx.x;
    if (e >= E_TOT) return;
    int d = (e < E_DIR) ? dst[e] : (e - E_DIR);
    atomicAdd(&deg[d], 1);
}

// single block, 1024 threads, each owns 64 nodes: exclusive prefix sum of degrees
__global__ void k_scan(const int* __restrict__ deg, int* __restrict__ row_ptr,
                       int* __restrict__ cursor) {
    __shared__ int s[1024];
    int t = threadIdx.x;
    int base = t * 64;
    int sum = 0;
    for (int i = 0; i < 64; i++) sum += deg[base + i];
    s[t] = sum;
    __syncthreads();
    for (int off = 1; off < 1024; off <<= 1) {
        int v = (t >= off) ? s[t - off] : 0;
        __syncthreads();
        s[t] += v;
        __syncthreads();
    }
    int run = (t == 0) ? 0 : s[t - 1];
    for (int i = 0; i < 64; i++) {
        row_ptr[base + i] = run;
        cursor[base + i] = run;
        run += deg[base + i];
    }
    if (t == 1023) row_ptr[N_NODES] = run;
}

__global__ void k_scatter(const int* __restrict__ src, const int* __restrict__ dst,
                          const float* __restrict__ ea, const float* __restrict__ msum,
                          int* __restrict__ cursor, int* __restrict__ col,
                          float* __restrict__ eaw) {
    int e = blockIdx.x * 256 + threadIdx.x;
    if (e >= E_TOT) return;
    int s, d; float v;
    if (e < E_DIR) { s = src[e]; d = dst[e]; v = ea[e]; }
    else           { s = d = e - E_DIR; v = msum[0] * (1.0f / E_DIR); }
    int pos = atomicAdd(&cursor[d], 1);
    col[pos] = s;
    eaw[pos] = v;
}

// ---------------- WMMA GEMM: xl = h@Wl + bl ; xr = h@Wr + br ----------------
// one wave per 16x16 output tile; A fragment reused for both B matrices
__global__ void k_gemm_xlxr(const _Float16* __restrict__ hh,
                            const _Float16* __restrict__ WlT,   // [128n][128k] f16
                            const _Float16* __restrict__ WrT,
                            const float* __restrict__ bl, const float* __restrict__ br,
                            float* __restrict__ xl, float* __restrict__ xr) {
    int wave = (int)((blockIdx.x * (size_t)blockDim.x + threadIdx.x) >> 5);
    int lane = threadIdx.x & 31;
    int mt = wave >> 3;            // 0..4095
    int nt = wave & 7;             // 0..7
    int m0 = mt << 4, n0 = nt << 4;
    int lr = lane & 15, hi = lane >> 4;

    const _Float16* aptr = hh  + (size_t)(m0 + lr) * HIDF + hi * 16;  // A: row-major h
    const _Float16* blp  = WlT + (size_t)(n0 + lr) * HIDF + hi * 16;  // B: transposed W
    const _Float16* brp  = WrT + (size_t)(n0 + lr) * HIDF + hi * 16;

    f32x8 cl = {}, cr = {};
#pragma unroll
    for (int k0 = 0; k0 < HIDF; k0 += 32) {
        half8 alo = *(const half8*)(aptr + k0);        // K = k0+koff .. +7
        half8 ahi = *(const half8*)(aptr + k0 + 8);    // K = k0+koff+8 .. +15
        half16 a = __builtin_shufflevector(alo, ahi, 0,1,2,3,4,5,6,7,8,9,10,11,12,13,14,15);
        half16 bL = *(const half16*)(blp + k0);        // contiguous K per lane
        half16 bR = *(const half16*)(brp + k0);
        cl = __builtin_amdgcn_wmma_f32_16x16x32_f16(false, a, false, bL, (short)0, cl, false, false);
        cr = __builtin_amdgcn_wmma_f32_16x16x32_f16(false, a, false, bR, (short)0, cr, false, false);
    }
    int col = lane & 15;
    float bLv = bl[n0 + col], bRv = br[n0 + col];
#pragma unroll
    for (int v = 0; v < 8; v++) {
        int r = m0 + v + (hi << 3);                    // C layout: row = v + 8*(lane/16)
        xl[(size_t)r * HIDF + n0 + col] = cl[v] + bLv;
        xr[(size_t)r * HIDF + n0 + col] = cr[v] + bRv;
    }
}

// ---------------- fused GATv2 aggregation: one wave per destination node ----------------
// lane = channel within head; 4 heads carried in registers; zero atomics.
__global__ void k_gat_fused(const int* __restrict__ row_ptr, const int* __restrict__ col,
                            const float* __restrict__ eaw,
                            const float* __restrict__ We, const float* __restrict__ att,
                            const float* __restrict__ xl, const float* __restrict__ xr,
                            const float* __restrict__ gb,
                            float* __restrict__ logit_s,
                            float* __restrict__ h, _Float16* __restrict__ hh) {
    int wave = (int)((blockIdx.x * (size_t)blockDim.x + threadIdx.x) >> 5);
    int lane = threadIdx.x & 31;
    if (wave >= N_NODES) return;
    int d = wave;
    int start = row_ptr[d], end = row_ptr[d + 1];

    float xrh[NHEAD], wev[NHEAD], atv[NHEAD];
#pragma unroll
    for (int hd = 0; hd < NHEAD; hd++) {
        xrh[hd] = xr[(size_t)d * HIDF + hd * CH + lane];
        wev[hd] = We[hd * CH + lane];
        atv[hd] = att[hd * CH + lane];
    }

    // pass A: logits + per-head running max (butterfly reduce -> uniform on all lanes)
    float mx[NHEAD] = {-3.402823466e38f, -3.402823466e38f, -3.402823466e38f, -3.402823466e38f};
    for (int pos = start; pos < end; pos++) {
        int s = col[pos];
        float eav = eaw[pos];
        const float* xls = xl + (size_t)s * HIDF;
#pragma unroll
        for (int hd = 0; hd < NHEAD; hd++) {
            float v = xls[hd * CH + lane] + xrh[hd] + eav * wev[hd];
            v = (v > 0.f) ? v : NEG_SLOPE * v;         // leaky_relu
            float p = atv[hd] * v;
            for (int off = 16; off; off >>= 1) p += __shfl_xor(p, off, 32);
            mx[hd] = fmaxf(mx[hd], p);
            if (lane == hd) logit_s[(size_t)pos * NHEAD + hd] = p;   // wave-owned slot
        }
    }

    // pass B: exp, denominator and 128-wide message fully in registers
    float den[NHEAD] = {0.f, 0.f, 0.f, 0.f};
    float msg[NHEAD] = {0.f, 0.f, 0.f, 0.f};
    for (int pos = start; pos < end; pos++) {
        int s = col[pos];
        const float* xls = xl + (size_t)s * HIDF;
#pragma unroll
        for (int hd = 0; hd < NHEAD; hd++) {
            float ex = __expf(logit_s[(size_t)pos * NHEAD + hd] - mx[hd]);
            den[hd] += ex;
            msg[hd] = fmaf(xls[hd * CH + lane], ex, msg[hd]);
        }
    }

    // normalize + bias + relu + residual, refresh f16 copy
#pragma unroll
    for (int hd = 0; hd < NHEAD; hd++) {
        int f = hd * CH + lane;
        size_t idx = (size_t)d * HIDF + f;
        float v = msg[hd] / den[hd] + gb[f];
        float nv = fmaxf(v, 0.f) + h[idx];
        h[idx] = nv;
        hh[idx] = (_Float16)nv;
    }
}

// ---------------- pooling: sum/max per (graph, feature) ----------------
__global__ void k_pool(const float* __restrict__ h, float* __restrict__ psum,
                       float* __restrict__ pmax) {
    int b = blockIdx.x, f = threadIdx.x;               // 64 blocks x 128 threads
    const float* base = h + (size_t)b * NPG * HIDF + f;
    float s = 0.f, mx = -3.402823466e38f;
    for (int i = 0; i < NPG; i++) {
        float v = base[(size_t)i * HIDF];
        s += v;
        mx = fmaxf(mx, v);
    }
    psum[b * HIDF + f] = s;
    pmax[b * HIDF + f] = mx;
}

// ---------------- fused action encoder + Q head: one block per (graph, action) ----------------
__global__ void k_actions(const float* __restrict__ h, const float* __restrict__ at,
                          const float* __restrict__ psum, const float* __restrict__ pmax,
                          const float* __restrict__ aW1, const float* __restrict__ ab1,
                          const float* __restrict__ aW2, const float* __restrict__ ab2,
                          const float* __restrict__ qW1, const float* __restrict__ qb1,
                          const float* __restrict__ qW2, const float* __restrict__ qb2,
                          const float* __restrict__ qW3, const float* __restrict__ qb3,
                          float* __restrict__ out) {
    __shared__ float ai[520];
    __shared__ float t0[128];
    __shared__ float t1[512];
    __shared__ float red[128];
    int ba = blockIdx.x;                               // 0..2047
    int b = ba >> 5;
    int tid = threadIdx.x;                             // 0..127
    const float* av = at + (size_t)ba * 7;
    // gather 4 node embeddings -> ai[0..511]
    for (int t = tid; t < 512; t += 128) {
        int j = t >> 7, f = t & 127;
        int ni = (int)av[j] + b * NPG;
        ai[t] = h[(size_t)ni * HIDF + f];
    }
    if (tid < 3) ai[512 + tid] = av[4 + tid];
    __syncthreads();
    // a1: 515 -> 128, relu
    float a1 = ab1[tid];
    for (int k = 0; k < MLP_IN; k++) a1 = fmaf(ai[k], aW1[k * HIDF + tid], a1);
    t0[tid] = fmaxf(a1, 0.f);
    __syncthreads();
    // a2: 128 -> 128, relu
    float a2 = ab2[tid];
    for (int k = 0; k < HIDF; k++) a2 = fmaf(t0[k], aW2[k * HIDF + tid], a2);
    float ae = fmaxf(a2, 0.f);
    // comb = [sum, mean, max, ae]
    float sm = psum[b * HIDF + tid];
    t1[tid] = sm;
    t1[128 + tid] = sm * (1.0f / NPG);
    t1[256 + tid] = pmax[b * HIDF + tid];
    t1[384 + tid] = ae;
    __syncthreads();
    // q1: 512 -> 128, relu
    float q1 = qb1[tid];
    for (int k = 0; k < 512; k++) q1 = fmaf(t1[k], qW1[k * HIDF + tid], q1);
    q1 = fmaxf(q1, 0.f);
    __syncthreads();
    t0[tid] = q1;
    __syncthreads();
    // q2: 128 -> 128, relu
    float q2 = qb2[tid];
    for (int k = 0; k < HIDF; k++) q2 = fmaf(t0[k], qW2[k * HIDF + tid], q2);
    q2 = fmaxf(q2, 0.f);
    // q3: 128 -> 1 (block reduce)
    red[tid] = q2 * qW3[tid];
    __syncthreads();
    for (int s = 64; s > 0; s >>= 1) {
        if (tid < s) red[tid] += red[tid + s];
        __syncthreads();
    }
    if (tid == 0) out[ba] = red[0] + qb3[0];
}

// =====================================================================

extern "C" void kernel_launch(void* const* d_in, const int* in_sizes, int n_in,
                              void* d_out, int out_size, void* d_ws, size_t ws_size,
                              hipStream_t stream) {
    const float* x           = (const float*)d_in[0];
    const float* edge_attr   = (const float*)d_in[1];
    const float* action_t    = (const float*)d_in[2];
    const int*   edge_index  = (const int*)d_in[3];
    // d_in[4] batch_idx, d_in[5] batch_ptr: structure is constant (b*1024)
    const float* Wn  = (const float*)d_in[6];
    const float* bn  = (const float*)d_in[7];
    const float* gWl = (const float*)d_in[8];
    const float* gWr = (const float*)d_in[9];
    const float* gWe = (const float*)d_in[10];
    const float* gatt= (const float*)d_in[11];
    const float* gbl = (const float*)d_in[12];
    const float* gbr = (const float*)d_in[13];
    const float* gb  = (const float*)d_in[14];
    const float* aW1 = (const float*)d_in[15];
    const float* ab1 = (const float*)d_in[16];
    const float* aW2 = (const float*)d_in[17];
    const float* ab2 = (const float*)d_in[18];
    const float* qW1 = (const float*)d_in[19];
    const float* qb1 = (const float*)d_in[20];
    const float* qW2 = (const float*)d_in[21];
    const float* qb2 = (const float*)d_in[22];
    const float* qW3 = (const float*)d_in[23];
    const float* qb3 = (const float*)d_in[24];
    float* out = (float*)d_out;

    const int* e_src = edge_index;
    const int* e_dst = edge_index + E_DIR;

    // --- carve workspace (256B aligned) ---
    char* p = (char*)d_ws;
    auto carve = [&](size_t bytes) -> char* {
        char* r = p;
        p += (bytes + 255) & ~(size_t)255;
        return r;
    };
    float*     h       = (float*)    carve((size_t)N_NODES * HIDF * 4);   // 32 MB
    _Float16*  hh      = (_Float16*) carve((size_t)N_NODES * HIDF * 2);   // 16 MB
    float*     xl      = (float*)    carve((size_t)N_NODES * HIDF * 4);   // 32 MB
    float*     xr      = (float*)    carve((size_t)N_NODES * HIDF * 4);   // 32 MB
    float*     logit_s = (float*)    carve((size_t)E_TOT * NHEAD * 4);    // 5.2 MB
    int*       deg     = (int*)      carve((size_t)N_NODES * 4);
    int*       row_ptr = (int*)      carve((size_t)(N_NODES + 1) * 4);
    int*       cursor  = (int*)      carve((size_t)N_NODES * 4);
    int*       csr_col = (int*)      carve((size_t)E_TOT * 4);
    float*     eaw     = (float*)    carve((size_t)E_TOT * 4);
    _Float16*  WlT     = (_Float16*) carve((size_t)LLAYERS * HIDF * HIDF * 2);
    _Float16*  WrT     = (_Float16*) carve((size_t)LLAYERS * HIDF * HIDF * 2);
    float*     psum    = (float*)    carve((size_t)BG * HIDF * 4);
    float*     pmax    = (float*)    carve((size_t)BG * HIDF * 4);
    float*     msum    = (float*)    carve(256);

    const int NH_ELEMS = 32768;                        // N*HID / 256
    const int EDGE_BLK = (E_TOT + 255) / 256;          // 1280

    // weight prep + node encode + edge-attr mean
    k_prep_w<<<(LLAYERS * HIDF * HIDF + 255) / 256, 256, 0, stream>>>(gWl, gWr, WlT, WrT);
    k_node_encode<<<NH_ELEMS, 256, 0, stream>>>(x, Wn, bn, h, hh);
    k_zero_scalar<<<1, 32, 0, stream>>>(msum);
    k_edge_sum<<<(E_DIR + 255) / 256, 256, 0, stream>>>(edge_attr, msum);

    // CSR-by-destination (reused by all 3 layers)
    k_zero_deg<<<(N_NODES + 255) / 256, 256, 0, stream>>>(deg);
    k_hist<<<EDGE_BLK, 256, 0, stream>>>(e_dst, deg);
    k_scan<<<1, 1024, 0, stream>>>(deg, row_ptr, cursor);
    k_scatter<<<EDGE_BLK, 256, 0, stream>>>(e_src, e_dst, edge_attr, msum,
                                            cursor, csr_col, eaw);

    for (int l = 0; l < LLAYERS; l++) {
        k_gemm_xlxr<<<4096, 256, 0, stream>>>(hh,
                                              WlT + (size_t)l * HIDF * HIDF,
                                              WrT + (size_t)l * HIDF * HIDF,
                                              gbl + l * HIDF, gbr + l * HIDF,
                                              xl, xr);
        k_gat_fused<<<(N_NODES + 7) / 8, 256, 0, stream>>>(row_ptr, csr_col, eaw,
                                                           gWe + l * HIDF,
                                                           gatt + l * NHEAD * CH,
                                                           xl, xr, gb + l * HIDF,
                                                           logit_s, h, hh);
    }

    k_pool<<<BG, HIDF, 0, stream>>>(h, psum, pmax);
    k_actions<<<BG * AACT, HIDF, 0, stream>>>(h, action_t, psum, pmax,
                                              aW1, ab1, aW2, ab2,
                                              qW1, qb1, qW2, qb2, qW3, qb3, out);
    (void)in_sizes; (void)n_in; (void)out_size; (void)ws_size;
}
